// StyleGAN3Generator_79817672229324
// MI455X (gfx1250) — matrix operand order repaired and checked by reading, not compiled, not run on hardware
//
#include <hip/hip_runtime.h>

// ---- MI455X / gfx1250 mapping-network kernel ----
// Fuses pixelnorm + 8 dense(512x512)+lrelu layers + x16 broadcast.
// f16 WMMA (V_WMMA_F32_16X16X32_F16) with f32 accumulate; wscale/bias applied in f32.

typedef _Float16 v16h __attribute__((ext_vector_type(16)));
typedef _Float16 v8h  __attribute__((ext_vector_type(8)));
typedef _Float16 v4h  __attribute__((ext_vector_type(4)));
typedef float    v8f  __attribute__((ext_vector_type(8)));
typedef float    v4f  __attribute__((ext_vector_type(4)));

#define HIDDEN 512
#define LAYERS 8
#define ROWS_PER_BLOCK 64
#define NOUT 16
#define WSCALE 4.4194173824159216e-4f  /* 0.01 / sqrt(512) */
#define BSCALE 0.01f
#define SQRT2  1.4142135623730951f

// Swizzle f32 weights -> f16 in WMMA B-matrix per-lane layout:
// sw[l][kt(16)][nt(32)][lane(32)][e(16)], element = W[n][k] with
// n = nt*16 + lane%16, k = kt*32 + (e%8) + 8*(lane/16) + 16*(e/8)  (matches 16-bit B layout).
__global__ void swizzle_weights(const float* __restrict__ w, _Float16* __restrict__ sw) {
  int idx = blockIdx.x * blockDim.x + threadIdx.x;   // 0 .. 8*16*32*32*16-1 (2M)
  int e    =  idx        & 15;
  int lane = (idx >> 4)  & 31;
  int nt   = (idx >> 9)  & 31;
  int kt   = (idx >> 14) & 15;
  int l    =  idx >> 18;
  int n = nt * 16 + (lane & 15);
  int k = kt * 32 + (e & 7) + ((lane >> 4) << 3) + ((e >> 3) << 4);
  sw[idx] = (_Float16)w[((size_t)l * HIDDEN + n) * HIDDEN + k];
}

__global__ void __launch_bounds__(256)
mapping_kernel(const float* __restrict__ z, const _Float16* __restrict__ sw,
               const float* __restrict__ bias, float* __restrict__ out) {
  extern __shared__ char smem[];                 // 64 KB: act[64][512] f16 (in-place per layer)
  _Float16* act  = (_Float16*)smem;
  float*    redf = (float*)smem;                 // transient: 256 partial sums (pre-activation)
  float*    sclf = (float*)(smem + 4096);        // transient: 64 row scales

  const int tid    = threadIdx.x;
  const int lane   = tid & 31;
  const int wv     = tid >> 5;
  const int waveM  = wv >> 2;        // 0..1 : which 32-row half
  const int waveN  = wv & 3;         // 0..3 : which 128-col slice
  const int laneLo = lane & 15;
  const int laneHi = lane >> 4;
  const size_t blockRow = (size_t)blockIdx.x * ROWS_PER_BLOCK;

  // ---------- pixel norm: z -> f16 activations in LDS ----------
  {
    const int row = tid >> 2;        // 0..63
    const int q   = tid & 3;         // 128-col chunk
    const float* zp = z + (blockRow + row) * HIDDEN + q * 128;
    float s = 0.f;
#pragma unroll
    for (int i = 0; i < 32; ++i) {
      v4f v = *(const v4f*)(zp + i * 4);
      s += v.x * v.x + v.y * v.y + v.z * v.z + v.w * v.w;
    }
    redf[tid] = s;
    __syncthreads();
    if (tid < 64) {
      float t = redf[tid * 4] + redf[tid * 4 + 1] + redf[tid * 4 + 2] + redf[tid * 4 + 3];
      sclf[tid] = rsqrtf(t * (1.0f / HIDDEN) + 1e-8f);
    }
    __syncthreads();
    const float sc = sclf[row];       // read scale into register...
    __syncthreads();                  // ...before activations overwrite the scratch
    _Float16* ap = act + row * HIDDEN + q * 128;
#pragma unroll
    for (int i = 0; i < 32; ++i) {
      v4f v = *(const v4f*)(zp + i * 4);
      v4h h;
      h.x = (_Float16)(v.x * sc); h.y = (_Float16)(v.y * sc);
      h.z = (_Float16)(v.z * sc); h.w = (_Float16)(v.w * sc);
      *(v4h*)(ap + i * 4) = h;
    }
    __syncthreads();
  }

  // A-tile row pointers (fixed: in-place activation buffer)
  const _Float16* arow0 = act + (waveM * 32 + laneLo) * HIDDEN;        // M-tile 0
  const _Float16* arow1 = act + (waveM * 32 + 16 + laneLo) * HIDDEN;   // M-tile 1

  const v8f vzero = {0.f, 0.f, 0.f, 0.f, 0.f, 0.f, 0.f, 0.f};
  v8f acc[2][8];

  for (int l = 0; l < LAYERS; ++l) {
#pragma unroll
    for (int mt = 0; mt < 2; ++mt)
#pragma unroll
      for (int nt = 0; nt < 8; ++nt) acc[mt][nt] = vzero;

    const _Float16* wl = sw + (size_t)l * (16 * 32 * 32 * 16);

#pragma unroll 2
    for (int kt = 0; kt < 16; ++kt) {
      // A tiles: 16x32 f16, per-lane = two contiguous 16B chunks from LDS
      const int colBase = kt * 32 + laneHi * 8;
      v8h lo0 = *(const v8h*)(arow0 + colBase);
      v8h hi0 = *(const v8h*)(arow0 + colBase + 16);
      v8h lo1 = *(const v8h*)(arow1 + colBase);
      v8h hi1 = *(const v8h*)(arow1 + colBase + 16);
      v16h a0 = __builtin_shufflevector(lo0, hi0, 0,1,2,3,4,5,6,7,8,9,10,11,12,13,14,15);
      v16h a1 = __builtin_shufflevector(lo1, hi1, 0,1,2,3,4,5,6,7,8,9,10,11,12,13,14,15);

      const _Float16* bp = wl + ((size_t)(kt * 32 + waveN * 8) * 32 + lane) * 16;
#pragma unroll
      for (int nt = 0; nt < 8; ++nt) {
        v16h b = *(const v16h*)(bp + nt * (32 * 16));   // contiguous 32B per lane
        acc[0][nt] = __builtin_amdgcn_wmma_f32_16x16x32_f16(
            false, a0, false, b, (short)0, acc[0][nt], false, false);
        acc[1][nt] = __builtin_amdgcn_wmma_f32_16x16x32_f16(
            false, a1, false, b, (short)0, acc[1][nt], false, false);
      }
    }

    __syncthreads();   // all waves done reading act before it is overwritten

    if (l < LAYERS - 1) {
      // epilogue: scale, bias, lrelu*sqrt2, f16 back to LDS (in place)
#pragma unroll
      for (int nt = 0; nt < 8; ++nt) {
        const int n  = waveN * 128 + nt * 16 + laneLo;
        const float bv = bias[l * HIDDEN + n] * BSCALE;
#pragma unroll
        for (int mt = 0; mt < 2; ++mt) {
          const int rowb = waveM * 32 + mt * 16 + laneHi * 8;
#pragma unroll
          for (int r = 0; r < 8; ++r) {
            float v = fmaf(acc[mt][nt][r], WSCALE, bv);
            v = (v > 0.f ? v : 0.2f * v) * SQRT2;
            act[(rowb + r) * HIDDEN + n] = (_Float16)v;
          }
        }
      }
      __syncthreads();
    } else {
      // final layer: write f32, broadcast x16 along the NUM_OUTPUTS axis (NT stores)
      float* op = out + blockRow * NOUT * HIDDEN;
#pragma unroll
      for (int nt = 0; nt < 8; ++nt) {
        const int n  = waveN * 128 + nt * 16 + laneLo;
        const float bv = bias[l * HIDDEN + n] * BSCALE;
#pragma unroll
        for (int mt = 0; mt < 2; ++mt) {
          const int rowb = waveM * 32 + mt * 16 + laneHi * 8;
#pragma unroll
          for (int r = 0; r < 8; ++r) {
            float v = fmaf(acc[mt][nt][r], WSCALE, bv);
            v = (v > 0.f ? v : 0.2f * v) * SQRT2;
            float* p = op + (size_t)(rowb + r) * NOUT * HIDDEN + n;
#pragma unroll
            for (int rep = 0; rep < NOUT; ++rep)
              __builtin_nontemporal_store(v, p + rep * HIDDEN);
          }
        }
      }
    }
  }
}

extern "C" void kernel_launch(void* const* d_in, const int* in_sizes, int n_in,
                              void* d_out, int out_size, void* d_ws, size_t ws_size,
                              hipStream_t stream) {
  const float* z    = (const float*)d_in[0];   // [16384, 512]
  const float* w    = (const float*)d_in[1];   // [8, 512, 512]
  const float* b    = (const float*)d_in[2];   // [8, 512]
  float* out        = (float*)d_out;           // [16384, 16, 512]
  _Float16* sw      = (_Float16*)d_ws;         // 4 MB swizzled f16 weights

  const int batch   = in_sizes[0] / HIDDEN;    // 16384
  const int nblocks = batch / ROWS_PER_BLOCK;  // 256

  // 8*16*32*32*16 = 2,097,152 elements -> 8192 blocks of 256
  swizzle_weights<<<8192, 256, 0, stream>>>(w, sw);
  mapping_kernel<<<nblocks, 256, 65536, stream>>>(z, sw, b, out);
}
